// refattention_27376121544779
// MI455X (gfx1250) — matrix operand order repaired and checked by reading, not compiled, not run on hardware
//
#include <hip/hip_runtime.h>

typedef __attribute__((ext_vector_type(16))) _Float16 v16h;
typedef __attribute__((ext_vector_type(8)))  _Float16 v8h;
typedef __attribute__((ext_vector_type(4)))  _Float16 v4h;
typedef __attribute__((ext_vector_type(8)))  float    v8f;
typedef __attribute__((ext_vector_type(4)))  float    v4f;

#define BB 8
#define CC 64
#define NN 2304          // 48*48
#define WAVES 4          // waves per workgroup, 16 query rows each
#define SROW 2320        // LDS score row stride in halfs (2304 + 16 pad)
#define SC_HALFS (WAVES*16*SROW)
#define VSTR 40          // V-stage row stride in halfs (32 data + 8 pad; 80B, 16B-aligned)

// ---------------------------------------------------------------------------
// q = (x + gamma*pos) / sqrt(sum_c(.^2) + 1e-6), stored f16 as [B][N][C]
// ---------------------------------------------------------------------------
__global__ void prep_q_kernel(const float* __restrict__ x,
                              const float* __restrict__ pos,
                              const float* __restrict__ gamma_p,
                              _Float16* __restrict__ qh) {
  int idx = blockIdx.x * blockDim.x + threadIdx.x;
  if (idx >= BB * NN) return;
  int b = idx / NN, n = idx - b * NN;
  float g = gamma_p[0];
  const float* xb = x   + (size_t)b * CC * NN + n;
  const float* pb = pos + (size_t)b * CC * NN + n;
  float s = 0.f;
  #pragma unroll 8
  for (int c = 0; c < CC; ++c) {
    float t = xb[(size_t)c * NN] + g * pb[(size_t)c * NN];
    s += t * t;
  }
  float inv = rsqrtf(s + 1e-6f);
  _Float16* qr = qh + (size_t)idx * CC;
  #pragma unroll 8
  for (int c = 0; c < CC; ++c) {
    float t = xb[(size_t)c * NN] + g * pb[(size_t)c * NN];
    qr[c] = (_Float16)(t * inv);
  }
}

// v (f32 [B][C][N]) -> f16 same layout
__global__ void conv_v_kernel(const float* __restrict__ v, _Float16* __restrict__ vh) {
  int i = blockIdx.x * blockDim.x + threadIdx.x;
  if (i < BB * CC * NN) vh[i] = (_Float16)v[i];
}

// ---------------------------------------------------------------------------
// Fused attention: per wave -> 16 query rows.
//   GEMM1 (K=64): S = Q_n . Q_m^T * beta_gamma  -> LDS (f16, full 2304 row)
//   softmax (vectorized v4h passes + shfl butterflies), beta -> global (NT)
//   GEMM2 (K=2304): O = expS @ V^T; V chunks staged in LDS via async-to-LDS
//   out = w0*inv*O + w1*v
// ---------------------------------------------------------------------------
__global__ __launch_bounds__(WAVES * 32)
void attn_kernel(const _Float16* __restrict__ qh,
                 const _Float16* __restrict__ vhp,
                 const float*    __restrict__ vf,
                 const float*    __restrict__ bg_p,
                 const float*    __restrict__ sg0_p,
                 const float*    __restrict__ sg1_p,
                 float* __restrict__ out,
                 float* __restrict__ beta) {
  __shared__ __align__(16) _Float16 vstage[2][CC * VSTR];   // 2 x 5120 B
  extern __shared__ __align__(16) char smem[];
  _Float16* sc    = (_Float16*)smem;                        // WAVES*16 score rows (f16)
  float*    inv_s = (float*)(smem + (size_t)SC_HALFS * 2);

  const int tile = blockIdx.x % (NN / (16 * WAVES));
  const int b    = blockIdx.x / (NN / (16 * WAVES));
  const int wave = threadIdx.x >> 5;
  const int lane = threadIdx.x & 31;
  const int n0   = tile * (16 * WAVES) + wave * 16;
  const int col  = lane & 15;      // A: row M, B: col N, C/D: col N
  const int hi   = lane >> 4;      // 0 or 1
  const int kb   = hi * 8;         // A-matrix K sub-base (16-bit A layout)
  const int rbase = hi * 8;        // C/D row base
  const float bg = bg_p[0];

  // ---- A = 16 q-rows [16 x 64] as WMMA A operand, contiguous k order (16*hi + s) (2 x v16h for K=0..63)
  const _Float16* qrow = qh + ((size_t)b * NN + n0 + col) * CC;
  v8h a00 = *(const v8h*)(qrow + hi * 16);
  v8h a01 = *(const v8h*)(qrow + hi * 16 + 8);
  v8h a10 = *(const v8h*)(qrow + hi * 16 + 32);
  v8h a11 = *(const v8h*)(qrow + hi * 16 + 40);
  v16h A0 = __builtin_shufflevector(a00, a01, 0,1,2,3,4,5,6,7,8,9,10,11,12,13,14,15);
  v16h A1 = __builtin_shufflevector(a10, a11, 0,1,2,3,4,5,6,7,8,9,10,11,12,13,14,15);

  // ---- GEMM1: scores -> LDS f16
  #pragma unroll 2
  for (int mt = 0; mt < NN / 16; ++mt) {
    const _Float16* qcol = qh + ((size_t)b * NN + mt * 16 + col) * CC + hi * 16;
    v16h B0 = *(const v16h*)qcol;
    v16h B1 = *(const v16h*)(qcol + 32);
    v8f s = {};
    s = __builtin_amdgcn_wmma_f32_16x16x32_f16(false, A0, false, B0, (short)0, s, false, false);
    s = __builtin_amdgcn_wmma_f32_16x16x32_f16(false, A1, false, B1, (short)0, s, false, false);
    #pragma unroll
    for (int i = 0; i < 8; ++i)
      sc[(size_t)(wave * 16 + rbase + i) * SROW + mt * 16 + col] = (_Float16)(s[i] * bg);
  }

  // ---- softmax (each wave owns its 16 rows), v4h-vectorized passes
  for (int r = 0; r < 16; ++r) {
    _Float16* row = sc + (size_t)(wave * 16 + r) * SROW;
    float mx = -3.0e38f;
    #pragma unroll 2
    for (int it = 0; it < NN / 128; ++it) {
      v4h d = *(const v4h*)(row + it * 128 + lane * 4);
      #pragma unroll
      for (int j = 0; j < 4; ++j) mx = fmaxf(mx, (float)d[j]);
    }
    #pragma unroll
    for (int off = 16; off > 0; off >>= 1)
      mx = fmaxf(mx, __shfl_xor(mx, off, 32));
    float sum = 0.f;
    #pragma unroll 2
    for (int it = 0; it < NN / 128; ++it) {
      int m4 = it * 128 + lane * 4;
      v4h d = *(const v4h*)(row + m4);
      v4h e;
      #pragma unroll
      for (int j = 0; j < 4; ++j) {
        float ev = __expf((float)d[j] - mx);
        e[j] = (_Float16)ev;               // keep unnormalized exp for GEMM2
        sum += ev;
      }
      *(v4h*)(row + m4) = e;
    }
    #pragma unroll
    for (int off = 16; off > 0; off >>= 1)
      sum += __shfl_xor(sum, off, 32);
    float inv = 1.f / sum;
    if (lane == 0) inv_s[wave * 16 + r] = inv;
    // beta[b, n0+r, :] — coalesced non-temporal v4 stores (write-once 170MB stream)
    float* bd = beta + (size_t)b * NN * NN + (size_t)(n0 + r) * NN;
    #pragma unroll 2
    for (int it = 0; it < NN / 128; ++it) {
      int m4 = it * 128 + lane * 4;
      v4h e4 = *(const v4h*)(row + m4);
      v4f f4;
      #pragma unroll
      for (int j = 0; j < 4; ++j) f4[j] = (float)e4[j] * inv;
      __builtin_nontemporal_store(f4, (v4f*)(bd + m4));
    }
  }

  // ---- GEMM2: O[16 x 64] = expS[16 x 2304] @ V^T, K in chunks of 32.
  // V chunk (4KB) staged in LDS by async-to-LDS copies, shared by all 4 waves,
  // double-buffered; ASYNCcnt + barrier for cross-wave visibility.
  const int t    = threadIdx.x;
  const int vc   = t >> 1;                 // c row handled by this thread
  const int vhh  = (t & 1) * 16;           // half-index within 32-half chunk row
  const _Float16* vsrc = vhp + ((size_t)b * CC + vc) * NN + vhh;
  unsigned lds0 = (unsigned)(unsigned long long)(uintptr_t)&vstage[0][vc * VSTR + vhh];
  unsigned lds1 = (unsigned)(unsigned long long)(uintptr_t)&vstage[1][vc * VSTR + vhh];

#define ISSUE_V(KT, LDSA)                                                        \
  { unsigned long long ga = (unsigned long long)(uintptr_t)(vsrc + (size_t)(KT) * 32); \
    asm volatile("global_load_async_to_lds_b128 %0, %1, off\n\t"                 \
                 "global_load_async_to_lds_b128 %0, %1, off offset:16"           \
                 :: "v"(LDSA), "v"(ga) : "memory"); }

  ISSUE_V(0, lds0);

  v8f z = {};
  v8f acc[4] = {z, z, z, z};
  const _Float16* arow = sc + (size_t)(wave * 16 + col) * SROW;   // A row = lane&15
  for (int kt = 0; kt < NN / 32; ++kt) {
    asm volatile("s_wait_asynccnt 0" ::: "memory");
    __syncthreads();                       // chunk kt visible to all waves
    if (kt + 1 < NN / 32) {                // prefetch next chunk into other buffer
      if ((kt + 1) & 1) ISSUE_V(kt + 1, lds1) else ISSUE_V(kt + 1, lds0)
    }
    v8h p0 = *(const v8h*)(arow + kt * 32 + hi * 16);
    v8h p1 = *(const v8h*)(arow + kt * 32 + hi * 16 + 8);
    v16h Ak = __builtin_shufflevector(p0, p1, 0,1,2,3,4,5,6,7,8,9,10,11,12,13,14,15);
    const _Float16* vs = &vstage[kt & 1][0];
    #pragma unroll
    for (int ct = 0; ct < 4; ++ct) {
      v16h Bk = *(const v16h*)(vs + (ct * 16 + col) * VSTR + hi * 16);
      acc[ct] = __builtin_amdgcn_wmma_f32_16x16x32_f16(false, Ak, false, Bk, (short)0,
                                                       acc[ct], false, false);
    }
  }
#undef ISSUE_V

  // ---- epilogue: out = w0*inv*O + w1*v   (out layout [B][C][N])
  float e0 = __expf(sg0_p[0]);
  float e1 = __expf(sg1_p[0]);
  float w0 = e0 / (e0 + e1), w1 = e1 / (e0 + e1);
  #pragma unroll
  for (int ct = 0; ct < 4; ++ct) {
    int c = ct * 16 + col;
    float*       op = out + ((size_t)b * CC + c) * NN + n0 + rbase;
    const float* vp = vf  + ((size_t)b * CC + c) * NN + n0 + rbase;
    #pragma unroll
    for (int i = 0; i < 8; ++i) {
      float inv = inv_s[wave * 16 + rbase + i];
      op[i] = w0 * inv * acc[ct][i] + w1 * vp[i];
    }
  }
}

// ---------------------------------------------------------------------------
extern "C" void kernel_launch(void* const* d_in, const int* in_sizes, int n_in,
                              void* d_out, int out_size, void* d_ws, size_t ws_size,
                              hipStream_t stream) {
  (void)in_sizes; (void)n_in; (void)out_size; (void)ws_size;
  const float* x   = (const float*)d_in[0];
  const float* v   = (const float*)d_in[1];
  const float* pos = (const float*)d_in[2];
  const float* gam = (const float*)d_in[3];
  const float* bg  = (const float*)d_in[4];
  const float* sg0 = (const float*)d_in[5];
  const float* sg1 = (const float*)d_in[6];

  float* out  = (float*)d_out;
  float* beta = out + (size_t)BB * CC * NN;          // outputs concatenated: out, beta

  _Float16* qh = (_Float16*)d_ws;                    // [B][N][C] f16  (2.36 MB)
  _Float16* vh = qh + (size_t)BB * NN * CC;          // [B][C][N] f16  (2.36 MB)

  prep_q_kernel<<<(BB * NN + 255) / 256, 256, 0, stream>>>(x, pos, gam, qh);
  conv_v_kernel<<<(BB * CC * NN + 255) / 256, 256, 0, stream>>>(v, vh);

  size_t shmem = (size_t)SC_HALFS * sizeof(_Float16) + WAVES * 16 * sizeof(float); // ~297 KB dynamic
  dim3 grid(BB * (NN / (16 * WAVES)));               // 8 * 36 = 288 workgroups
  attn_kernel<<<grid, WAVES * 32, shmem, stream>>>(qh, vh, v, bg, sg0, sg1, out, beta);
}
